// OneScalePredictor_2078764172046
// MI455X (gfx1250) — matrix operand order repaired and checked by reading, not compiled, not run on hardware
//
#include <hip/hip_runtime.h>
#include <hip/hip_bf16.h>
#include <stdint.h>

typedef __attribute__((ext_vector_type(16))) _Float16 v16h;
typedef __attribute__((ext_vector_type(8)))  _Float16 v8h;
typedef __attribute__((ext_vector_type(8)))  float    v8f;

#define ROWS_PER_BLOCK 128   // 8 wave32 * 16 rows
#define THREADS 256
#define C64 64

// ---------------- LDS layout (element offsets) ----------------
// f16 region: weights stored PRE-SWIZZLED into WMMA B-fragment order:
//   fragment (nt,kt) occupies 512 halves at ((nt*ksteps)+kt)*512,
//   within a fragment: lane*16 + e  ->  W[kt*32 + (lane>=16)*16 + e][nt*16 + lane%16]
#define OFF_WD1   0          // 64x64   (ksteps=2, ntiles=4)
#define OFF_WD2   4096
#define OFF_WP1   8192
#define OFF_WP2   12288      // 64x256 (255 padded)
#define OFF_WU1   28672      // 96x64  (72 padded, ksteps=3)
#define OFF_WU2   34816
#define OFF_WU3   38912
#define OFF_WU4   43008      // 64x512
#define OFF_ACTX  75776      // 128x64
#define OFF_ACTT  83968      // 128x64
#define OFF_ACTF  92160      // 128x96
#define OFF_ACTU  104448     // 128x64
#define OFF_ACTV  112640     // 128x64
#define HALF_TOTAL 120832
// f32 region
#define FOFF_BD1  0
#define FOFF_AD   64
#define FOFF_BD2  128
#define FOFF_BP1  192
#define FOFF_AP   256
#define FOFF_BP2  320        // 256
#define FOFF_BU1  576
#define FOFF_AU1  640
#define FOFF_BU2  704
#define FOFF_AU2  768
#define FOFF_BU3  832
#define FOFF_BU4  896        // 512
#define FLOAT_TOTAL 1408
// i32 region: idx buffer 128 rows x 8 children
#define INT_TOTAL 1024

#define LDS_BYTES (HALF_TOTAL * 2 + FLOAT_TOTAL * 4 + INT_TOTAL * 4)

// ---- weight staging: global f32 row-major -> LDS f16 fragment-swizzled ----
__device__ __forceinline__ void stage_w_swz(_Float16* dst, const float* __restrict__ src,
                                            int Krows, int Ncols, int Kpad, int Npad, int tid) {
    const int ksteps = Kpad >> 5;
    const int total  = Kpad * Npad;
    for (int i = tid; i < total; i += THREADS) {
        const int frag = i >> 9;             // 512 halves per fragment
        const int lo   = i & 511;
        const int ln   = lo >> 4;            // lane 0..31
        const int e    = lo & 15;
        const int nt   = frag / ksteps;
        const int kt   = frag - nt * ksteps;
        const int row  = kt * 32 + (ln >> 4) * 16 + e;
        const int col  = nt * 16 + (ln & 15);
        float v = 0.f;
        if (row < Krows && col < Ncols) v = src[row * Ncols + col];
        dst[i] = (_Float16)v;
    }
}

// ---- A fragment (16x32 f16): two contiguous 16B chunks per lane -> 2x ds_load_b128 ----
__device__ __forceinline__ v16h lds_a_frag(const _Float16* act, int ldr, int k0, int lane) {
    const int hf = (lane >> 4) & 1;
    const int mr = lane & 15;
    const _Float16* p = act + mr * ldr + k0 + hf * 16;
    v8h lo = *(const v8h*)p;          // K +0..7   (relative)
    v8h hi = *(const v8h*)(p + 8);    // K +8..15  (relative)
    return __builtin_shufflevector(lo, hi, 0,1,2,3,4,5,6,7,8,9,10,11,12,13,14,15);
}

// ---- B fragment from pre-swizzled LDS: 32B contiguous per lane -> 2x ds_load_b128 ----
__device__ __forceinline__ v16h lds_b_frag(const _Float16* wswz, int ksteps, int nt, int kt, int lane) {
    const _Float16* p = wswz + (((nt * ksteps) + kt) << 9) + (lane << 4);
    v8h lo = *(const v8h*)p;
    v8h hi = *(const v8h*)(p + 8);
    return __builtin_shufflevector(lo, hi, 0,1,2,3,4,5,6,7,8,9,10,11,12,13,14,15);
}

// ---- generic 16-row tile GEMM, branch-free epilogue via template flags ----
template<int KSTEPS, bool ALPHA, bool RESID>
__device__ __forceinline__ void gemm_tile(const _Float16* actIn, int ldi,
                                          const _Float16* Wswz, int nouts,
                                          const float* bias, const float* alpha,
                                          const _Float16* resid, int ldres,
                                          _Float16* out, int ldo, int lane)
{
    const int hf = (lane >> 4) & 1;
    const int nc = lane & 15;
    for (int nt = 0; nt < (nouts >> 4); ++nt) {
        const int n0 = nt * 16;
        const float bv = bias[n0 + nc];
        const float al = ALPHA ? alpha[n0 + nc] : 0.f;
        v8f acc = {bv, bv, bv, bv, bv, bv, bv, bv};
#pragma unroll
        for (int kt = 0; kt < KSTEPS; ++kt) {
            v16h a = lds_a_frag(actIn, ldi, kt * 32, lane);
            v16h b = lds_b_frag(Wswz, KSTEPS, nt, kt, lane);
            acc = __builtin_amdgcn_wmma_f32_16x16x32_f16(false, a, false, b,
                                                         (short)0, acc, false, false);
        }
#pragma unroll
        for (int r = 0; r < 8; ++r) {
            const int row = r + 8 * hf;
            float v = acc[r];
            if (RESID) v += (float)resid[row * ldres + n0 + nc];
            if (ALPHA) v = (v >= 0.f) ? v : al * v;
            out[row * ldo + n0 + nc] = (_Float16)v;
        }
    }
}

// ---------------- Kernel A: invert the static gather ----------------
__global__ void build_inv_kernel(const int* __restrict__ sel, int* __restrict__ inv, int M) {
    const int i = blockIdx.x * THREADS + threadIdx.x;
    if (i < M) inv[sel[i]] = i;
}

// ---------------- Kernel B: fused one-scale predictor ----------------
__global__ void __launch_bounds__(THREADS)
fused_predictor_kernel(const float* __restrict__ X,      // [N,64]
                       const int*   __restrict__ bin,    // [N,8]
                       const int*   __restrict__ inv,    // [N*8]
                       const float* __restrict__ Wd1, const float* __restrict__ bd1,
                       const float* __restrict__ ad,
                       const float* __restrict__ Wd2, const float* __restrict__ bd2,
                       const float* __restrict__ Wp1, const float* __restrict__ bp1,
                       const float* __restrict__ ap,
                       const float* __restrict__ Wp2, const float* __restrict__ bp2,
                       const float* __restrict__ Wu1, const float* __restrict__ bu1,
                       const float* __restrict__ au1,
                       const float* __restrict__ Wu2, const float* __restrict__ bu2,
                       const float* __restrict__ au2,
                       const float* __restrict__ Wu3, const float* __restrict__ bu3,
                       const float* __restrict__ Wu4, const float* __restrict__ bu4,
                       float* __restrict__ newF,         // [M,64]
                       float* __restrict__ pred,         // [N,255]
                       int16_t* __restrict__ oct,        // [N]
                       int nNodes)
{
    extern __shared__ char smemraw[];
    _Float16* sh = (_Float16*)smemraw;
    float*    sf = (float*)(smemraw + HALF_TOTAL * 2);
    int*      si = (int*)(smemraw + HALF_TOTAL * 2 + FLOAT_TOTAL * 4);

    const int tid  = threadIdx.x;
    const int wave = tid >> 5;
    const int lane = tid & 31;
    const int hf   = (lane >> 4) & 1;
    const int nc   = lane & 15;
    const int row0g = blockIdx.x * ROWS_PER_BLOCK + wave * 16;

    // Prefetch next block's activation tile into L2 (global_prefetch_b8)
    if (blockIdx.x + 1 < gridDim.x && tid < 64) {
        const float* nxt = X + (size_t)(blockIdx.x + 1) * ROWS_PER_BLOCK * C64 + (size_t)tid * 512;
        __builtin_prefetch(nxt, 0, 1);
    }

    // ---- stage weights into fragment-swizzled f16 LDS (block-cooperative) ----
    stage_w_swz(sh + OFF_WD1, Wd1, 64, 64,  64, 64,  tid);
    stage_w_swz(sh + OFF_WD2, Wd2, 64, 64,  64, 64,  tid);
    stage_w_swz(sh + OFF_WP1, Wp1, 64, 64,  64, 64,  tid);
    stage_w_swz(sh + OFF_WP2, Wp2, 64, 255, 64, 256, tid);
    stage_w_swz(sh + OFF_WU1, Wu1, 72, 64,  96, 64,  tid);
    stage_w_swz(sh + OFF_WU2, Wu2, 64, 64,  64, 64,  tid);
    stage_w_swz(sh + OFF_WU3, Wu3, 64, 64,  64, 64,  tid);
    stage_w_swz(sh + OFF_WU4, Wu4, 64, 512, 64, 512, tid);
    if (tid < 64) {
        sf[FOFF_BD1 + tid] = bd1[tid];  sf[FOFF_AD  + tid] = ad[tid];
        sf[FOFF_BD2 + tid] = bd2[tid];  sf[FOFF_BP1 + tid] = bp1[tid];
        sf[FOFF_AP  + tid] = ap[tid];   sf[FOFF_BU1 + tid] = bu1[tid];
        sf[FOFF_AU1 + tid] = au1[tid];  sf[FOFF_BU2 + tid] = bu2[tid];
        sf[FOFF_AU2 + tid] = au2[tid];  sf[FOFF_BU3 + tid] = bu3[tid];
    }
    for (int i = tid; i < 256; i += THREADS) sf[FOFF_BP2 + i] = (i < 255) ? bp2[i] : 0.f;
    for (int i = tid; i < 512; i += THREADS) sf[FOFF_BU4 + i] = bu4[i];

    // ---- per-wave staging ----
    _Float16* actX = sh + OFF_ACTX + wave * 16 * 64;
    _Float16* actT = sh + OFF_ACTT + wave * 16 * 64;
    _Float16* actF = sh + OFF_ACTF + wave * 16 * 96;
    _Float16* actU = sh + OFF_ACTU + wave * 16 * 64;
    _Float16* actV = sh + OFF_ACTV + wave * 16 * 64;
    int*      idxW = si + wave * 128;

#pragma unroll 4
    for (int r = 0; r < 16; ++r) {
        const float2 v = ((const float2*)X)[(size_t)(row0g + r) * 32 + lane];
        actX[r * 64 + lane * 2]     = (_Float16)v.x;
        actX[r * 64 + lane * 2 + 1] = (_Float16)v.y;
    }
    for (int i = lane; i < 128; i += 32) {                 // 16 rows x 8 children
        const int lr = i >> 3, c = i & 7;
        const int b = bin[(size_t)(row0g + lr) * 8 + c];
        idxW[i] = b ? inv[(size_t)(row0g + lr) * 8 + c] : -1;
        actF[lr * 96 + 64 + c] = (_Float16)(float)(b << 8);
    }
    for (int i = lane; i < 16 * 24; i += 32) {             // zero-pad cols 72..95
        const int lr = i / 24, c = 72 + i % 24;
        actF[lr * 96 + c] = (_Float16)0.f;
    }
    if (lane < 16) {                                       // octree code
        int s = 0;
#pragma unroll
        for (int k = 0; k < 8; ++k) s += bin[(size_t)(row0g + lane) * 8 + k] << k;
        oct[row0g + lane] = (int16_t)(s - 1);
    }
    __syncthreads();

    // ---- dec residual block ----
    gemm_tile<2, true,  false>(actX, 64, sh + OFF_WD1, 64, sf + FOFF_BD1, sf + FOFF_AD,
                               nullptr, 0, actT, 64, lane);                 // h
    __syncthreads();
    gemm_tile<2, false, true >(actT, 64, sh + OFF_WD2, 64, sf + FOFF_BD2, nullptr,
                               actX, 64, actF, 96, lane);                   // rec -> f[:,0:64]
    __syncthreads();

    // ---- pred branch ----
    gemm_tile<2, true,  false>(actF, 96, sh + OFF_WP1, 64, sf + FOFF_BP1, sf + FOFF_AP,
                               nullptr, 0, actT, 64, lane);                 // p
    __syncthreads();
    for (int nt = 0; nt < 16; ++nt) {                      // cur_pred = p @ Wp2 + bp2
        const int n0 = nt * 16;
        const float bv = sf[FOFF_BP2 + n0 + nc];
        v8f acc = {bv, bv, bv, bv, bv, bv, bv, bv};
#pragma unroll
        for (int kt = 0; kt < 2; ++kt) {
            v16h a = lds_a_frag(actT, 64, kt * 32, lane);
            v16h b = lds_b_frag(sh + OFF_WP2, 2, nt, kt, lane);
            acc = __builtin_amdgcn_wmma_f32_16x16x32_f16(false, a, false, b,
                                                         (short)0, acc, false, false);
        }
        const int col = n0 + nc;
        if (col < 255) {
#pragma unroll
            for (int r = 0; r < 8; ++r)
                pred[(size_t)(row0g + r + 8 * hf) * 255 + col] = acc[r];
        }
    }
    __syncthreads();

    // ---- upsample chain ----
    gemm_tile<3, true,  false>(actF, 96, sh + OFF_WU1, 64, sf + FOFF_BU1, sf + FOFF_AU1,
                               nullptr, 0, actU, 64, lane);                 // u
    __syncthreads();
    gemm_tile<2, true,  false>(actU, 64, sh + OFF_WU2, 64, sf + FOFF_BU2, sf + FOFF_AU2,
                               nullptr, 0, actT, 64, lane);                 // h2
    __syncthreads();
    gemm_tile<2, false, true >(actT, 64, sh + OFF_WU3, 64, sf + FOFF_BU3, nullptr,
                               actU, 64, actV, 64, lane);                   // u + h2@Wu3
    __syncthreads();

    // ---- up = u @ Wu4 (8 children x 64), fused child scatter ----
    for (int c = 0; c < 8; ++c) {
        for (int t = 0; t < 4; ++t) {
            const int nt = c * 4 + t;
            const int n0 = nt * 16;
            const float bv = sf[FOFF_BU4 + n0 + nc];
            v8f acc = {bv, bv, bv, bv, bv, bv, bv, bv};
#pragma unroll
            for (int kt = 0; kt < 2; ++kt) {
                v16h a = lds_a_frag(actV, 64, kt * 32, lane);
                v16h b = lds_b_frag(sh + OFF_WU4, 2, nt, kt, lane);
                acc = __builtin_amdgcn_wmma_f32_16x16x32_f16(false, a, false, b,
                                                             (short)0, acc, false, false);
            }
#pragma unroll
            for (int r = 0; r < 8; ++r) {
                const int lr = r + 8 * hf;
                const int j = idxW[lr * 8 + c];
                if (j >= 0) newF[(size_t)j * 64 + t * 16 + nc] = acc[r];
            }
        }
    }
}

extern "C" void kernel_launch(void* const* d_in, const int* in_sizes, int n_in,
                              void* d_out, int out_size, void* d_ws, size_t ws_size,
                              hipStream_t stream) {
    const float* X   = (const float*)d_in[0];
    const int*   bin = (const int*)d_in[1];
    const int*   sel = (const int*)d_in[2];
    const float *Wd1 = (const float*)d_in[3],  *bd1 = (const float*)d_in[4],
                *ad  = (const float*)d_in[5],
                *Wd2 = (const float*)d_in[6],  *bd2 = (const float*)d_in[7],
                *Wp1 = (const float*)d_in[8],  *bp1 = (const float*)d_in[9],
                *ap  = (const float*)d_in[10],
                *Wp2 = (const float*)d_in[11], *bp2 = (const float*)d_in[12],
                *Wu1 = (const float*)d_in[13], *bu1 = (const float*)d_in[14],
                *au1 = (const float*)d_in[15],
                *Wu2 = (const float*)d_in[16], *bu2 = (const float*)d_in[17],
                *au2 = (const float*)d_in[18],
                *Wu3 = (const float*)d_in[19], *bu3 = (const float*)d_in[20],
                *Wu4 = (const float*)d_in[21], *bu4 = (const float*)d_in[22];

    const int nNodes = in_sizes[0] / C64;
    const int M      = in_sizes[2];

    float*   newF = (float*)d_out;
    float*   pred = newF + (size_t)M * C64;
    int16_t* oct  = (int16_t*)(pred + (size_t)nNodes * 255);
    int*     inv  = (int*)d_ws;   // [nNodes*8]

    build_inv_kernel<<<(M + THREADS - 1) / THREADS, THREADS, 0, stream>>>(sel, inv, M);

    hipFuncSetAttribute((const void*)fused_predictor_kernel,
                        hipFuncAttributeMaxDynamicSharedMemorySize, LDS_BYTES);

    const int blocks = nNodes / ROWS_PER_BLOCK;
    fused_predictor_kernel<<<blocks, THREADS, LDS_BYTES, stream>>>(
        X, bin, inv,
        Wd1, bd1, ad, Wd2, bd2,
        Wp1, bp1, ap, Wp2, bp2,
        Wu1, bu1, au1, Wu2, bu2, au2, Wu3, bu3, Wu4, bu4,
        newF, pred, oct, nNodes);
}
